// GPT_79302276153555
// MI455X (gfx1250) — matrix operand order repaired and checked by reading, not compiled, not run on hardware
//
#include <hip/hip_runtime.h>
#include <hip/hip_bf16.h>

// ---- problem constants (GPT-2 small fwd) ----
#define Bn 4
#define Tn 1024
#define Ln 12
#define Hn 12
#define Cn 768
#define Vn 50257
#define Mn (Bn * Tn) /* 4096 rows of the activation matrix */

typedef __attribute__((ext_vector_type(16))) __bf16 v16bf;
typedef __attribute__((ext_vector_type(8)))  float  v8f;
typedef unsigned int       u32;
typedef unsigned long long u64;
typedef __attribute__((ext_vector_type(4))) u32 u32x4;
typedef __attribute__((ext_vector_type(8))) int i32x8;
typedef __attribute__((ext_vector_type(4))) int i32x4;

#if __has_builtin(__builtin_amdgcn_tensor_load_to_lds)
#define HAVE_TDM 1
#else
#define HAVE_TDM 0
#endif

__device__ __forceinline__ v8f wmma_bf16(v16bf a, v16bf b, v8f c) {
    // D = A(16x32 bf16) * B(32x16 bf16) + C(16x16 f32)
    return __builtin_amdgcn_wmma_f32_16x16x32_bf16(false, a, false, b,
                                                   (short)0, c, false, false);
}

__device__ __forceinline__ float gelu_tanh(float v) {
    const float k0 = 0.7978845608028654f; // sqrt(2/pi)
    const float k1 = 0.044715f;
    float u = k0 * (v + k1 * v * v * v);
    return 0.5f * v * (1.0f + tanhf(u));
}

// A-fragment K index for element i, lane-half h (contiguous order: k = 16*h + i, matches B fills)
__device__ __forceinline__ int a_kidx(int i, int h) {
    return i + 16 * h;
}

#if HAVE_TDM
// ---------------------------------------------------------------------------
// TDM 2-D tile load: tile_d1 rows of tile_d0 bf16 elements, global row stride
// stride0 elements, zero-fill rows >= tens_d1 (OOB), optional LDS row padding.
// D# bit layout per CDNA5 ISA.
// 6-arg builtin (amdgpu-toolchain clang-23): (g0, g1, g2, g3, g_extra, cpol).
// ---------------------------------------------------------------------------
__device__ __forceinline__ void tdm_load_2d(u32 lds_off, const void* gptr,
                                            u32 tile_d0, u32 tile_d1,
                                            u32 tens_d0, u32 tens_d1,
                                            u64 stride0,
                                            u32 pad_en, u32 pad_int, u32 pad_amt) {
    u64 ga = (u64)gptr;
    u32x4 g0;
    g0[0] = 1u;                                   // count=1, user mode
    g0[1] = lds_off;                              // lds_addr (bytes)
    g0[2] = (u32)(ga & 0xffffffffu);              // global_addr[31:0]
    g0[3] = (u32)((ga >> 32) & 0x01ffffffu)       // global_addr[56:32]
          | (2u << 30);                           // type=2 (image)
    i32x8 g1;
    g1[0] = (int)((1u << 16)                      // data_size=1 -> 2 bytes
          | (pad_en << 20) | (pad_int << 22) | (pad_amt << 25));
    g1[1] = (int)((tens_d0 & 0xffffu) << 16);     // tensor_dim0[15:0]
    g1[2] = (int)((tens_d0 >> 16) | ((tens_d1 & 0xffffu) << 16));
    g1[3] = (int)((tens_d1 >> 16) | (tile_d0 << 16));
    g1[4] = (int)(tile_d1 & 0xffffu);             // tile_dim1, tile_dim2=0
    g1[5] = (int)(stride0 & 0xffffffffu);         // tensor_dim0_stride[31:0]
    g1[6] = (int)((stride0 >> 32) & 0xffffu);     // stride[47:32], dim1_stride=0
    g1[7] = 0;
    i32x4 gz4 = {0, 0, 0, 0};
    i32x8 gz8 = {0, 0, 0, 0, 0, 0, 0, 0};
    __builtin_amdgcn_tensor_load_to_lds(g0, g1, gz4, gz4, gz8, 0);
}
#endif

// ---------------------------------------------------------------------------
// 1. fp32 -> bf16 convert (weights, once per launch — deterministic)
// ---------------------------------------------------------------------------
__global__ __launch_bounds__(256) void cvt_bf16_kernel(const float* __restrict__ in,
                                                       __bf16* __restrict__ out,
                                                       size_t n) {
    size_t i = (size_t)blockIdx.x * 256 + threadIdx.x;
    if (i < n) out[i] = (__bf16)in[i];
}

// ---------------------------------------------------------------------------
// 2. embedding: x[b,t,:] = wte[idx[b,t],:] + wpe[t,:]
// ---------------------------------------------------------------------------
__global__ __launch_bounds__(256) void embed_kernel(const int* __restrict__ idx,
                                                    const float* __restrict__ wte,
                                                    const float* __restrict__ wpe,
                                                    float* __restrict__ x) {
    int bt  = blockIdx.x;
    int t   = bt % Tn;
    int tok = idx[bt];
    for (int i = threadIdx.x; i < Cn; i += 256)
        x[(size_t)bt * Cn + i] = wte[(size_t)tok * Cn + i] + wpe[(size_t)t * Cn + i];
}

// ---------------------------------------------------------------------------
// 3. layernorm: fp32 row -> bf16 row
// ---------------------------------------------------------------------------
__global__ __launch_bounds__(256) void layernorm_kernel(const float* __restrict__ x,
                                                        const float* __restrict__ g,
                                                        const float* __restrict__ b,
                                                        __bf16* __restrict__ out) {
    __shared__ float red[256];
    int row = blockIdx.x;
    const float* xr = x + (size_t)row * Cn;
    float s = 0.f, s2 = 0.f;
    for (int i = threadIdx.x; i < Cn; i += 256) {
        float v = xr[i];
        s += v; s2 += v * v;
    }
    red[threadIdx.x] = s; __syncthreads();
    for (int off = 128; off >= 1; off >>= 1) {
        if (threadIdx.x < off) red[threadIdx.x] += red[threadIdx.x + off];
        __syncthreads();
    }
    float mean = red[0] / (float)Cn;
    __syncthreads();
    red[threadIdx.x] = s2; __syncthreads();
    for (int off = 128; off >= 1; off >>= 1) {
        if (threadIdx.x < off) red[threadIdx.x] += red[threadIdx.x + off];
        __syncthreads();
    }
    float var = red[0] / (float)Cn - mean * mean;
    float inv = rsqrtf(var + 1e-5f);
    for (int i = threadIdx.x; i < Cn; i += 256)
        out[(size_t)row * Cn + i] = (__bf16)((xr[i] - mean) * inv * g[i] + b[i]);
}

// ---------------------------------------------------------------------------
// 4. bf16 WMMA GEMM: Out[M,N] = epi(A[M,K] @ W[N,K]^T + bias)
//    128 threads (4 waves), 64x64 block tile, 32x32 per wave, K-step 32.
//    A tile 64x32 [m][k], B tile 64x32 [n][k] (both row stride 34 bf16),
//    TDM-staged, double-buffered.
//    EPI: 0 = fp32, 1 = fp32 + residual, 2 = gelu -> bf16, 3 = bf16
// ---------------------------------------------------------------------------
#define GEMM_TILE_BYTES (64 * 34 * 2)                 /* 4352 */
#define GEMM_BUF_BYTES  (2 * GEMM_TILE_BYTES)         /* A+B   */
#define GEMM_LDS_BYTES  (2 * GEMM_BUF_BYTES)          /* x2 buffers */

template <int EPI>
__global__ __launch_bounds__(128)
void gemm_bf16_kernel(const __bf16* __restrict__ A, const __bf16* __restrict__ W,
                      const float* __restrict__ bias, const float* __restrict__ resid,
                      float* __restrict__ outf, __bf16* __restrict__ outb,
                      int M, int N, int K) {
    extern __shared__ char smem[];

    const int tid  = threadIdx.x;
    const int lane = tid & 31;
    const int wave = tid >> 5;
    const int wm   = (wave >> 1) * 32;
    const int wn   = (wave & 1) * 32;
    const int half = lane >> 4;
    const int l16  = lane & 15;
    const int m0   = blockIdx.y * 64;
    const int n0   = blockIdx.x * 64;
    const u32 nrem = (u32)(N - n0) < 64u ? (u32)(N - n0) : 64u;

    v8f acc[2][2] = {};
    const int nk = K / 32;

#if HAVE_TDM
    const u32 lds_base = __builtin_amdgcn_groupstaticsize();
    // issue tile 0
    if (wave == 0) {
        tdm_load_2d(lds_base + 0 * GEMM_BUF_BYTES, A + (size_t)m0 * K,
                    32, 64, 32, 64, (u64)K, 1, 3, 0);
        tdm_load_2d(lds_base + 0 * GEMM_BUF_BYTES + GEMM_TILE_BYTES, W + (size_t)n0 * K,
                    32, 64, 32, nrem, (u64)K, 1, 3, 0);
    }
#endif

    for (int it = 0; it < nk; ++it) {
        const int buf = it & 1;
        const __bf16* As  = (const __bf16*)(smem + buf * GEMM_BUF_BYTES);
        const __bf16* Bsn = (const __bf16*)(smem + buf * GEMM_BUF_BYTES + GEMM_TILE_BYTES);

#if HAVE_TDM
        if (wave == 0) {
            if (it + 1 < nk) {
                int k1 = (it + 1) * 32;
                int nb = (it + 1) & 1;
                tdm_load_2d(lds_base + nb * GEMM_BUF_BYTES, A + (size_t)m0 * K + k1,
                            32, 64, 32, 64, (u64)K, 1, 3, 0);
                tdm_load_2d(lds_base + nb * GEMM_BUF_BYTES + GEMM_TILE_BYTES,
                            W + (size_t)n0 * K + k1, 32, 64, 32, nrem, (u64)K, 1, 3, 0);
                __builtin_amdgcn_s_wait_tensorcnt(2);   // oldest tile (it) done
            } else {
                __builtin_amdgcn_s_wait_tensorcnt(0);
            }
        }
        __syncthreads();
#else
        // manual staging fallback
        {
            __bf16* Aw = (__bf16*)(smem + buf * GEMM_BUF_BYTES);
            __bf16* Bw = (__bf16*)(smem + buf * GEMM_BUF_BYTES + GEMM_TILE_BYTES);
            int k0 = it * 32;
#pragma unroll
            for (int i = 0; i < 16; ++i) {
                int e = i * 128 + tid;
                int r = e >> 5, c = e & 31;
                Aw[r * 34 + c] = (m0 + r < M) ? A[(size_t)(m0 + r) * K + k0 + c]
                                              : (__bf16)0.0f;
                Bw[r * 34 + c] = ((u32)r < nrem) ? W[(size_t)(n0 + r) * K + k0 + c]
                                                 : (__bf16)0.0f;
            }
        }
        __syncthreads();
#endif

        v16bf af[2], bf[2];
#pragma unroll
        for (int mt = 0; mt < 2; ++mt) {
            int row = wm + mt * 16 + l16;
#pragma unroll
            for (int i = 0; i < 16; ++i) af[mt][i] = As[row * 34 + a_kidx(i, half)];
        }
#pragma unroll
        for (int nt = 0; nt < 2; ++nt) {
            int col = wn + nt * 16 + l16;
#pragma unroll
            for (int i = 0; i < 16; ++i) bf[nt][i] = Bsn[col * 34 + i + 16 * half];
        }
#pragma unroll
        for (int mt = 0; mt < 2; ++mt)
#pragma unroll
            for (int nt = 0; nt < 2; ++nt)
                acc[mt][nt] = wmma_bf16(af[mt], bf[nt], acc[mt][nt]);
        __syncthreads();
    }

    // epilogue: C layout — vgpr r holds (m = r + 8*half, n = l16)
#pragma unroll
    for (int mt = 0; mt < 2; ++mt)
#pragma unroll
        for (int nt = 0; nt < 2; ++nt)
#pragma unroll
            for (int r = 0; r < 8; ++r) {
                int gm = m0 + wm + mt * 16 + r + 8 * half;
                int gn = n0 + wn + nt * 16 + l16;
                if (gm >= M || gn >= N) continue;
                float v = acc[mt][nt][r];
                if (bias) v += bias[gn];
                size_t o = (size_t)gm * N + gn;
                if (EPI == 0) {
                    outf[o] = v;
                } else if (EPI == 1) {
                    outf[o] = resid[o] + v;
                } else if (EPI == 2) {
                    outb[o] = (__bf16)gelu_tanh(v);
                } else {
                    outb[o] = (__bf16)v;
                }
            }
}

// ---------------------------------------------------------------------------
// 5. flash attention: 1 wave per 16 query rows of one (b,h).
//    K/V chunks (32 kv rows x 64 d, row stride 3C) staged by TDM,
//    double-buffered. S = Q K^T via WMMA, online softmax, Y += P V via WMMA.
// ---------------------------------------------------------------------------
#define ATT_KV_TILE_BYTES (32 * 72 * 2)               /* 4608 */
#define ATT_BUF_BYTES     (2 * ATT_KV_TILE_BYTES)     /* K+V  */
#define ATT_P_OFF         (2 * ATT_BUF_BYTES)         /* 18432 */
#define ATT_LDS_BYTES     (ATT_P_OFF + 16 * 40 * 2)   /* +1280 */

__global__ __launch_bounds__(32)
void attention_kernel(const __bf16* __restrict__ qkv, __bf16* __restrict__ y) {
    extern __shared__ char smem[];

    const int bh   = blockIdx.x;          // b*H + h
    const int b    = bh / Hn;
    const int h    = bh % Hn;
    const int q0   = blockIdx.y * 16;
    const int lane = threadIdx.x;
    const int half = lane >> 4;
    const int l16  = lane & 15;
    const float scale = 0.125f;           // 1/sqrt(64)

    const __bf16* kbase = qkv + (size_t)b * Tn * 3 * Cn + Cn + h * 64;
    const __bf16* vbase = qkv + (size_t)b * Tn * 3 * Cn + 2 * Cn + h * 64;

    // Q fragments (two K-steps over d = 0..31, 32..63)
    v16bf qf[2];
    {
        int t = q0 + l16;
        const __bf16* qrow = qkv + (size_t)(b * Tn + t) * (3 * Cn) + h * 64;
#pragma unroll
        for (int ks = 0; ks < 2; ++ks)
#pragma unroll
            for (int i = 0; i < 16; ++i)
                qf[ks][i] = qrow[ks * 32 + a_kidx(i, half)];
    }

    float rm[8], rl[8];
    v8f accy[4] = {};
#pragma unroll
    for (int r = 0; r < 8; ++r) { rm[r] = -1e30f; rl[r] = 0.f; }

    const int nchunks = (q0 + 15) / 32 + 1;   // causal: kv chunks of 32

#if HAVE_TDM
    const u32 lds_base = __builtin_amdgcn_groupstaticsize();
    // pad: 64 bf16 = 32 DW rows -> pad_interval code 4 (32 DW), pad 4 DW -> row 72
    tdm_load_2d(lds_base + 0 * ATT_BUF_BYTES, kbase,
                64, 32, 64, 32, (u64)(3 * Cn), 1, 4, 3);
    tdm_load_2d(lds_base + 0 * ATT_BUF_BYTES + ATT_KV_TILE_BYTES, vbase,
                64, 32, 64, 32, (u64)(3 * Cn), 1, 4, 3);
#endif

    for (int c = 0; c < nchunks; ++c) {
        const int buf = c & 1;
        const __bf16* Kl = (const __bf16*)(smem + buf * ATT_BUF_BYTES);
        const __bf16* Vl = (const __bf16*)(smem + buf * ATT_BUF_BYTES + ATT_KV_TILE_BYTES);
        __bf16* Pl = (__bf16*)(smem + ATT_P_OFF);
        int kv0 = c * 32;

#if HAVE_TDM
        __syncthreads();   // drain outstanding LDS reads of the buffer we overwrite
        if (c + 1 < nchunks) {
            int nb = (c + 1) & 1;
            size_t roff = (size_t)(kv0 + 32) * 3 * Cn;
            tdm_load_2d(lds_base + nb * ATT_BUF_BYTES, kbase + roff,
                        64, 32, 64, 32, (u64)(3 * Cn), 1, 4, 3);
            tdm_load_2d(lds_base + nb * ATT_BUF_BYTES + ATT_KV_TILE_BYTES, vbase + roff,
                        64, 32, 64, 32, (u64)(3 * Cn), 1, 4, 3);
            __builtin_amdgcn_s_wait_tensorcnt(2);   // chunk c resident
        } else {
            __builtin_amdgcn_s_wait_tensorcnt(0);
        }
#else
        {
            __bf16* Kw = (__bf16*)(smem + buf * ATT_BUF_BYTES);
            __bf16* Vw = (__bf16*)(smem + buf * ATT_BUF_BYTES + ATT_KV_TILE_BYTES);
            const __bf16* krow = kbase + (size_t)(kv0 + lane) * 3 * Cn;
            const __bf16* vrow = vbase + (size_t)(kv0 + lane) * 3 * Cn;
#pragma unroll
            for (int i = 0; i < 8; ++i) {
                ((uint4*)&Kw[lane * 72])[i] = ((const uint4*)krow)[i];
                ((uint4*)&Vw[lane * 72])[i] = ((const uint4*)vrow)[i];
            }
        }
        __syncthreads();
#endif

        // S tiles: S[16q x 16kv] x2 (kv 0..15, 16..31 of chunk)
        v8f sf[2] = {};
#pragma unroll
        for (int nt = 0; nt < 2; ++nt)
#pragma unroll
            for (int ks = 0; ks < 2; ++ks) {
                v16bf bfr;
#pragma unroll
                for (int i = 0; i < 16; ++i)
                    bfr[i] = Kl[(nt * 16 + l16) * 72 + ks * 32 + i + 16 * half];
                sf[nt] = wmma_bf16(qf[ks], bfr, sf[nt]);
            }

        // scale + causal mask + online softmax update
        float p[2][8];
#pragma unroll
        for (int r = 0; r < 8; ++r) {
            int tq  = q0 + r + 8 * half;
            float s0 = sf[0][r] * scale;
            float s1 = sf[1][r] * scale;
            if (kv0 + l16 > tq)      s0 = -1e30f;
            if (kv0 + 16 + l16 > tq) s1 = -1e30f;
            float mx = fmaxf(s0, s1);
#pragma unroll
            for (int off = 8; off >= 1; off >>= 1) mx = fmaxf(mx, __shfl_xor(mx, off));
            float mnew  = fmaxf(rm[r], mx);
            float alpha = __expf(rm[r] - mnew);
            float p0 = __expf(s0 - mnew);
            float p1 = __expf(s1 - mnew);
            float sum = p0 + p1;
#pragma unroll
            for (int off = 8; off >= 1; off >>= 1) sum += __shfl_xor(sum, off);
            rl[r] = rl[r] * alpha + sum;
            rm[r] = mnew;
#pragma unroll
            for (int dt = 0; dt < 4; ++dt) accy[dt][r] *= alpha;
            p[0][r] = p0; p[1][r] = p1;
        }

        // P (C layout) -> LDS -> A-fragment layout
        __syncthreads();
#pragma unroll
        for (int nt = 0; nt < 2; ++nt)
#pragma unroll
            for (int r = 0; r < 8; ++r)
                Pl[(r + 8 * half) * 40 + nt * 16 + l16] = (__bf16)p[nt][r];
        __syncthreads();

        v16bf pf;
#pragma unroll
        for (int i = 0; i < 16; ++i) pf[i] = Pl[l16 * 40 + a_kidx(i, half)];

        // Y[16 x 64] += P[16 x 32] @ V[32 x 64]
#pragma unroll
        for (int dt = 0; dt < 4; ++dt) {
            v16bf vf;
#pragma unroll
            for (int i = 0; i < 16; ++i)
                vf[i] = Vl[(i + 16 * half) * 72 + dt * 16 + l16];
            accy[dt] = wmma_bf16(pf, vf, accy[dt]);
        }
    }

    // normalize and store y (bf16, feeds attn-proj GEMM)
#pragma unroll
    for (int dt = 0; dt < 4; ++dt)
#pragma unroll
        for (int r = 0; r < 8; ++r) {
            int t = q0 + r + 8 * half;
            int d = dt * 16 + l16;
            float v = accy[dt][r] / rl[r];
            y[(size_t)(b * Tn + t) * Cn + h * 64 + d] = (__bf16)v;
        }
}

// ---------------------------------------------------------------------------
// host orchestration
// ---------------------------------------------------------------------------
static inline size_t align256(size_t x) { return (x + 255) & ~(size_t)255; }

extern "C" void kernel_launch(void* const* d_in, const int* in_sizes, int n_in,
                              void* d_out, int out_size, void* d_ws, size_t ws_size,
                              hipStream_t stream) {
    (void)in_sizes; (void)n_in; (void)out_size; (void)ws_size;

    const int*   idx   = (const int*)d_in[0];
    const float* wte   = (const float*)d_in[1];
    const float* wpe   = (const float*)d_in[2];
    const float* ln1g  = (const float*)d_in[3];
    const float* ln1b  = (const float*)d_in[4];
    const float* wqkv  = (const float*)d_in[5];
    const float* bqkv  = (const float*)d_in[6];
    const float* wattn = (const float*)d_in[7];
    const float* battn = (const float*)d_in[8];
    const float* ln2g  = (const float*)d_in[9];
    const float* ln2b  = (const float*)d_in[10];
    const float* wfc   = (const float*)d_in[11];
    const float* bfc   = (const float*)d_in[12];
    const float* wmlp  = (const float*)d_in[13];
    const float* bmlp  = (const float*)d_in[14];
    const float* lnfg  = (const float*)d_in[15];
    const float* lnfb  = (const float*)d_in[16];
    const float* wlm   = (const float*)d_in[17];

    const size_t nWqkv = (size_t)Ln * 3 * Cn * Cn;
    const size_t nWatt = (size_t)Ln * Cn * Cn;
    const size_t nWfc  = (size_t)Ln * 4 * Cn * Cn;
    const size_t nWmlp = (size_t)Ln * Cn * 4 * Cn;
    const size_t nWlm  = (size_t)Vn * Cn;

    char* ws  = (char*)d_ws;
    size_t off = 0;
    auto carve = [&](size_t bytes) { char* p = ws + off; off += align256(bytes); return p; };

    __bf16* wqkv_b = (__bf16*)carve(nWqkv * 2);
    __bf16* watt_b = (__bf16*)carve(nWatt * 2);
    __bf16* wfc_b  = (__bf16*)carve(nWfc * 2);
    __bf16* wmlp_b = (__bf16*)carve(nWmlp * 2);
    __bf16* wlm_b  = (__bf16*)carve(nWlm * 2);
    float*  xbuf   = (float*)carve((size_t)Mn * Cn * 4);
    __bf16* hbuf   = (__bf16*)carve((size_t)Mn * Cn * 2);
    __bf16* qkvbuf = (__bf16*)carve((size_t)Mn * 3 * Cn * 2);
    __bf16* ybuf   = (__bf16*)carve((size_t)Mn * Cn * 2);
    __bf16* fcbuf  = (__bf16*)carve((size_t)Mn * 4 * Cn * 2);

    auto cvt = [&](const float* src, __bf16* dst, size_t n) {
        cvt_bf16_kernel<<<(unsigned)((n + 255) / 256), 256, 0, stream>>>(src, dst, n);
    };
    cvt(wqkv,  wqkv_b, nWqkv);
    cvt(wattn, watt_b, nWatt);
    cvt(wfc,   wfc_b,  nWfc);
    cvt(wmlp,  wmlp_b, nWmlp);
    cvt(wlm,   wlm_b,  nWlm);

    embed_kernel<<<Mn, 256, 0, stream>>>(idx, wte, wpe, xbuf);

    const dim3 gemm_blk(128);
    auto ggrid = [](int M, int N) { return dim3((N + 63) / 64, (M + 63) / 64); };

    for (int l = 0; l < Ln; ++l) {
        const __bf16* Wq = wqkv_b + (size_t)l * 3 * Cn * Cn;
        const __bf16* Wa = watt_b + (size_t)l * Cn * Cn;
        const __bf16* Wf = wfc_b  + (size_t)l * 4 * Cn * Cn;
        const __bf16* Wm = wmlp_b + (size_t)l * Cn * 4 * Cn;

        layernorm_kernel<<<Mn, 256, 0, stream>>>(xbuf, ln1g + l * Cn, ln1b + l * Cn, hbuf);
        gemm_bf16_kernel<3><<<ggrid(Mn, 3 * Cn), gemm_blk, GEMM_LDS_BYTES, stream>>>(
            hbuf, Wq, bqkv + (size_t)l * 3 * Cn, nullptr, nullptr, qkvbuf, Mn, 3 * Cn, Cn);
        attention_kernel<<<dim3(Bn * Hn, Tn / 16), 32, ATT_LDS_BYTES, stream>>>(qkvbuf, ybuf);
        gemm_bf16_kernel<1><<<ggrid(Mn, Cn), gemm_blk, GEMM_LDS_BYTES, stream>>>(
            ybuf, Wa, battn + (size_t)l * Cn, xbuf, xbuf, nullptr, Mn, Cn, Cn);
        layernorm_kernel<<<Mn, 256, 0, stream>>>(xbuf, ln2g + l * Cn, ln2b + l * Cn, hbuf);
        gemm_bf16_kernel<2><<<ggrid(Mn, 4 * Cn), gemm_blk, GEMM_LDS_BYTES, stream>>>(
            hbuf, Wf, bfc + (size_t)l * 4 * Cn, nullptr, nullptr, fcbuf, Mn, 4 * Cn, Cn);
        gemm_bf16_kernel<1><<<ggrid(Mn, Cn), gemm_blk, GEMM_LDS_BYTES, stream>>>(
            fcbuf, Wm, bmlp + (size_t)l * Cn, xbuf, xbuf, nullptr, Mn, Cn, 4 * Cn);
    }

    layernorm_kernel<<<Mn, 256, 0, stream>>>(xbuf, lnfg, lnfb, hbuf);
    gemm_bf16_kernel<0><<<ggrid(Mn, Vn), gemm_blk, GEMM_LDS_BYTES, stream>>>(
        hbuf, wlm_b, nullptr, nullptr, (float*)d_out, nullptr, Mn, Vn, Cn);
}
